// MusicTransformerDecoderLayer_52828097741088
// MI455X (gfx1250) — matrix operand order repaired and checked by reading, not compiled, not run on hardware
//
#include <hip/hip_runtime.h>

typedef __attribute__((ext_vector_type(16))) __bf16 v16bf;
typedef __attribute__((ext_vector_type(8)))  __bf16 v8bf;
typedef __attribute__((ext_vector_type(8)))  float  v8f;

#define S_LEN 2048
#define DM    1024
#define DFF   4096
#define NB    2
#define NH    16
#define DH    64
#define MROWS (NB * S_LEN)   // 4096 token rows

// ---------------------------------------------------------------------------
// CDNA5 async global->LDS copy (ASYNCcnt) helpers
// ---------------------------------------------------------------------------
__device__ __forceinline__ void async_ld_b128(void* lds, const void* gaddr) {
  // per-lane: LDS[lds] = MEM[gaddr], 16 bytes, tracked by ASYNCcnt
  unsigned l = (unsigned)(uintptr_t)lds;  // low 32 bits of generic ptr = LDS offset
  asm volatile("global_load_async_to_lds_b128 %0, %1, off"
               :
               : "v"(l), "v"(gaddr)
               : "memory");
}
__device__ __forceinline__ void wait_async0() {
  asm volatile("s_wait_asynccnt 0x0" ::: "memory");
}

// ---------------------------------------------------------------------------
// WMMA helpers (bf16 A/B, f32 accumulate, 16x16x32)
// ---------------------------------------------------------------------------
__device__ __forceinline__ v8f wmma_bf16(v16bf a, v16bf b, v8f c) {
  return __builtin_amdgcn_wmma_f32_16x16x32_bf16(false, a, false, b, (short)0, c,
                                                 false, false);
}

// A-matrix 16x32 fragment from row-major LDS tile.
__device__ __forceinline__ v16bf frag_a16x32(const __bf16* p, int stride, int m0,
                                             int k0, int lane) {
  const int r = m0 + (lane & 15);
  const int c = k0 + ((lane >> 4) << 3);
  const __bf16* q = p + r * stride + c;
  v8bf lo = *reinterpret_cast<const v8bf*>(q);
  v8bf hi = *reinterpret_cast<const v8bf*>(q + 16);
  v16bf out;
#pragma unroll
  for (int i = 0; i < 8; ++i) { out[i] = lo[i]; out[i + 8] = hi[i]; }
  return out;
}

// B-matrix 32x16 fragment from "Bt" (N-major) LDS tile: row = N, cols = K.
__device__ __forceinline__ v16bf frag_b32x16(const __bf16* p, int stride, int n0,
                                             int k0, int lane) {
  const int r = n0 + (lane & 15);
  const int c = k0 + ((lane >> 4) << 3);
  const __bf16* q = p + r * stride + c;
  v8bf lo = *reinterpret_cast<const v8bf*>(q);
  v8bf hi = *reinterpret_cast<const v8bf*>(q + 16);
  v16bf out;
#pragma unroll
  for (int i = 0; i < 8; ++i) { out[i] = lo[i]; out[i + 8] = hi[i]; }
  return out;
}

// T5 one-sided log bucket (N_BUCKETS=32, MAX_DIST=128, max_exact=16)
__device__ __forceinline__ int rel_bucket(int rel) {
  int d = rel > 0 ? rel : 0;
  if (d < 16) return d;
  float ratio = __logf((float)d * (1.0f / 16.0f)) * (1.0f / 2.0794415416798357f);
  int lg = 16 + (int)(ratio * 16.0f);
  return lg < 31 ? lg : 31;
}

// ---------------------------------------------------------------------------
// fp32 [K,N] -> bf16 [N,K] transpose (tiled)
// ---------------------------------------------------------------------------
__global__ __launch_bounds__(256) void transpose_f32_to_bf16(
    const float* __restrict__ w, __bf16* __restrict__ wt, int K, int N) {
  __shared__ float tile[16][17];
  const int k0 = blockIdx.x * 16;
  const int n0 = blockIdx.y * 16;
  const int tx = threadIdx.x & 15;
  const int ty = threadIdx.x >> 4;
  tile[ty][tx] = w[(size_t)(k0 + ty) * N + n0 + tx];
  __syncthreads();
  wt[(size_t)(n0 + ty) * K + k0 + tx] = (__bf16)tile[tx][ty];
}

// ---------------------------------------------------------------------------
// V [B*S, H*Dh] bf16 -> Vt [B,H,Dh,S] bf16 (coalesced tiled transpose)
// ---------------------------------------------------------------------------
__global__ __launch_bounds__(256) void transpose_v_bf16(
    const __bf16* __restrict__ v, __bf16* __restrict__ vt) {
  __shared__ __bf16 tile[16][17];
  const int bh = blockIdx.z;  // b*NH + h
  const int b = bh >> 4;
  const int h = bh & 15;
  const int s0 = blockIdx.x * 16;
  const int d0 = blockIdx.y * 16;
  const int tx = threadIdx.x & 15;
  const int ty = threadIdx.x >> 4;
  tile[ty][tx] = v[(size_t)(b * S_LEN + s0 + ty) * DM + h * DH + d0 + tx];
  __syncthreads();
  vt[(size_t)(bh * DH + d0 + ty) * S_LEN + s0 + tx] = tile[tx][ty];
}

// ---------------------------------------------------------------------------
// LayerNorm over D=1024, fp32 in -> bf16 out. One 256-thread block per row.
// ---------------------------------------------------------------------------
__global__ __launch_bounds__(256) void layernorm_to_bf16(
    const float* __restrict__ x, const float* __restrict__ g,
    const float* __restrict__ b, __bf16* __restrict__ out) {
  __shared__ float red[256];
  const int row = blockIdx.x;
  const int t = threadIdx.x;
  const float* xr = x + (size_t)row * DM;
  float vals[4];
  float s = 0.0f;
#pragma unroll
  for (int i = 0; i < 4; ++i) { vals[i] = xr[t + i * 256]; s += vals[i]; }
  red[t] = s;
  __syncthreads();
  for (int o = 128; o > 0; o >>= 1) {
    if (t < o) red[t] += red[t + o];
    __syncthreads();
  }
  const float mu = red[0] * (1.0f / DM);
  __syncthreads();
  float s2 = 0.0f;
#pragma unroll
  for (int i = 0; i < 4; ++i) { float d = vals[i] - mu; s2 += d * d; }
  red[t] = s2;
  __syncthreads();
  for (int o = 128; o > 0; o >>= 1) {
    if (t < o) red[t] += red[t + o];
    __syncthreads();
  }
  const float rstd = rsqrtf(red[0] * (1.0f / DM) + 1e-5f);
  __bf16* orow = out + (size_t)row * DM;
#pragma unroll
  for (int i = 0; i < 4; ++i) {
    const int c = t + i * 256;
    orow[c] = (__bf16)((vals[i] - mu) * rstd * g[c] + b[c]);
  }
}

// ---------------------------------------------------------------------------
// Generic bf16 WMMA GEMM: C = A[M,K] @ Bt[N,K]^T + bias
// Block tile 128x128, K-step 32, 8 waves (wave tile 32x64),
// async global->LDS staging with double buffering.
// EPI 0: bias -> bf16   | 1: bias+GELU -> bf16
// EPI 2: res + val*mask -> f32 | 3: (res + val)*mask -> f32
// ---------------------------------------------------------------------------
template <int EPI>
__global__ __launch_bounds__(256) void gemm_bf16_wmma(
    const __bf16* __restrict__ A, const __bf16* __restrict__ Bt,
    const float* __restrict__ bias, const float* __restrict__ residual,
    const int* __restrict__ rowmask, __bf16* __restrict__ outb,
    float* __restrict__ outf, int M, int N, int K) {
  __shared__ __bf16 As[2][128 * 40];
  __shared__ __bf16 Bs[2][128 * 40];
  const int tid = threadIdx.x;
  const int lane = tid & 31;
  const int wave = tid >> 5;
  const int wm = wave & 3;   // 4 waves along M
  const int wn = wave >> 2;  // 2 waves along N
  const int m0 = blockIdx.y * 128;
  const int n0 = blockIdx.x * 128;

  v8f acc[2][4];
#pragma unroll
  for (int i = 0; i < 2; ++i)
#pragma unroll
    for (int j = 0; j < 4; ++j) acc[i][j] = {};

  const int srow = tid >> 1;
  const int sh16 = (tid & 1) * 16;

  // issue cols 0-7,16-23 of the 128x32 A and Bt tiles: 1 async b128 each per thread
  auto issue = [&](int k0, int buf) {
    async_ld_b128(&As[buf][srow * 40 + sh16],
                  A + (size_t)(m0 + srow) * K + k0 + sh16);
    async_ld_b128(&Bs[buf][srow * 40 + sh16],
                  Bt + (size_t)(n0 + srow) * K + k0 + sh16);
  };

  const int nk = K / 32;
  issue(0, 0);
  for (int t = 0; t < nk; ++t) {
    wait_async0();      // this wave's tile-t transfers done
    __syncthreads();    // everyone's transfers done / prev compute done
    if (t + 1 < nk) issue((t + 1) * 32, (t + 1) & 1);  // overlap with compute

    const __bf16* as = As[t & 1];
    const __bf16* bs = Bs[t & 1];
    v16bf af[2], bfr[4];
#pragma unroll
    for (int i = 0; i < 2; ++i) af[i] = frag_a16x32(as, 40, wm * 32 + i * 16, 0, lane);
#pragma unroll
    for (int j = 0; j < 4; ++j) bfr[j] = frag_b32x16(bs, 40, wn * 64 + j * 16, 0, lane);
#pragma unroll
    for (int i = 0; i < 2; ++i)
#pragma unroll
      for (int j = 0; j < 4; ++j) acc[i][j] = wmma_bf16(af[i], bfr[j], acc[i][j]);
  }

  const int hl = lane >> 4;
  const int cl = lane & 15;
#pragma unroll
  for (int i = 0; i < 2; ++i) {
#pragma unroll
    for (int j = 0; j < 4; ++j) {
#pragma unroll
      for (int v = 0; v < 8; ++v) {
        const int m = m0 + wm * 32 + i * 16 + v + 8 * hl;
        const int n = n0 + wn * 64 + j * 16 + cl;
        float val = acc[i][j][v] + bias[n];
        const size_t idx = (size_t)m * N + n;
        if (EPI == 0) {
          outb[idx] = (__bf16)val;
        } else if (EPI == 1) {
          val = 0.5f * val * (1.0f + erff(val * 0.70710678118654752f));
          outb[idx] = (__bf16)val;
        } else if (EPI == 2) {
          outf[idx] = residual[idx] + val * (float)rowmask[m];
        } else {
          outf[idx] = (residual[idx] + val) * (float)rowmask[m];
        }
      }
    }
  }
}

// ---------------------------------------------------------------------------
// Flash attention with rel-pos bias + causal + key-pad mask.
// One 128-thread (4 wave) block per (b, h, 64-query tile); wave = 16 queries.
// K and pre-transposed-V tiles double-buffered via async global->LDS.
// ---------------------------------------------------------------------------
__global__ __launch_bounds__(128) void attn_flash_wmma(
    const __bf16* __restrict__ Qb, const __bf16* __restrict__ Kb,
    const __bf16* __restrict__ Vt, const float* __restrict__ rel_bias,
    const int* __restrict__ amask, __bf16* __restrict__ att) {
  __shared__ __bf16 Qs[64 * 72];
  __shared__ __bf16 Ks[2][64 * 72];
  __shared__ __bf16 Vts[2][64 * 72];  // [d][key] (source pre-transposed)
  __shared__ __bf16 Ps[4 * 16 * 72];  // per-wave P strips

  const int bh = blockIdx.y;
  const int b = bh >> 4;
  const int h = bh & 15;
  const int q0 = blockIdx.x * 64;
  const int tid = threadIdx.x;
  const int lane = tid & 31;
  const int wave = tid >> 5;
  const int hl = lane >> 4;
  const int cl = lane & 15;

  // 64x64 bf16 tile stage: 128 threads x 4 async b128
  const int srow = tid >> 1;
  const int sc0 = (tid & 1) * 32;
  auto stage = [&](__bf16* lds, const __bf16* g, int gstride) {
#pragma unroll
    for (int i = 0; i < 4; ++i)
      async_ld_b128(&lds[srow * 72 + sc0 + i * 8],
                    g + (size_t)srow * gstride + sc0 + i * 8);
  };
  auto issueKV = [&](int kt, int buf) {
    stage(Ks[buf], Kb + (size_t)(b * S_LEN + kt * 64) * DM + h * DH, DM);
    stage(Vts[buf], Vt + (size_t)(bh * DH) * S_LEN + kt * 64, S_LEN);
  };

  stage(Qs, Qb + (size_t)(b * S_LEN + q0) * DM + h * DH, DM);
  issueKV(0, 0);

  v16bf aq0, aq1;
  v8f accO[4];
#pragma unroll
  for (int t = 0; t < 4; ++t) accO[t] = {};
  float mi[8], li[8];
#pragma unroll
  for (int v = 0; v < 8; ++v) { mi[v] = -1e30f; li[v] = 0.0f; }

  const float scale = 0.125f;      // 1/sqrt(64)
  const int nkt = blockIdx.x + 1;  // causal: key tiles 0..q-tile
  for (int kt = 0; kt < nkt; ++kt) {
    wait_async0();
    __syncthreads();
    if (kt == 0) {  // Q ready now (uniform branch)
      aq0 = frag_a16x32(Qs, 72, wave * 16, 0, lane);
      aq1 = frag_a16x32(Qs, 72, wave * 16, 32, lane);
    }
    if (kt + 1 < nkt) issueKV(kt + 1, (kt + 1) & 1);

    const __bf16* ks = Ks[kt & 1];
    const __bf16* vs = Vts[kt & 1];

    // scores: S = Q @ K^T
    v8f sc[4];
#pragma unroll
    for (int nt = 0; nt < 4; ++nt) {
      v8f s = {};
      v16bf bk0 = frag_b32x16(ks, 72, nt * 16, 0, lane);
      v16bf bk1 = frag_b32x16(ks, 72, nt * 16, 32, lane);
      s = wmma_bf16(aq0, bk0, s);
      s = wmma_bf16(aq1, bk1, s);
      sc[nt] = s;
    }

    // scale + rel-pos bias + causal/pad mask
#pragma unroll
    for (int nt = 0; nt < 4; ++nt) {
      const int j = kt * 64 + nt * 16 + cl;
      const int mp = amask[b * S_LEN + j];
#pragma unroll
      for (int v = 0; v < 8; ++v) {
        const int qrow = q0 + wave * 16 + v + 8 * hl;
        float val = sc[nt][v] * scale + rel_bias[rel_bucket(qrow - j) * NH + h];
        if (j > qrow || mp == 0) val = -1e30f;
        sc[nt][v] = val;
      }
    }

    // online softmax: each row lives in a 16-lane half-group at fixed vgpr idx
#pragma unroll
    for (int v = 0; v < 8; ++v) {
      float x = fmaxf(fmaxf(sc[0][v], sc[1][v]), fmaxf(sc[2][v], sc[3][v]));
      x = fmaxf(x, __shfl_xor(x, 1, 32));
      x = fmaxf(x, __shfl_xor(x, 2, 32));
      x = fmaxf(x, __shfl_xor(x, 4, 32));
      x = fmaxf(x, __shfl_xor(x, 8, 32));
      const float mnew = fmaxf(mi[v], x);
      const float corr = __expf(mi[v] - mnew);
      float sum = 0.0f;
#pragma unroll
      for (int nt = 0; nt < 4; ++nt) {
        const float p = __expf(sc[nt][v] - mnew);
        sc[nt][v] = p;
        sum += p;
      }
      sum += __shfl_xor(sum, 1, 32);
      sum += __shfl_xor(sum, 2, 32);
      sum += __shfl_xor(sum, 4, 32);
      sum += __shfl_xor(sum, 8, 32);
      li[v] = li[v] * corr + sum;
      mi[v] = mnew;
#pragma unroll
      for (int t = 0; t < 4; ++t) accO[t][v] = accO[t][v] * corr;
    }

    // P -> per-wave LDS strip (C->A relayout); same-wave LDS is in-order
    __bf16* pw = &Ps[wave * 16 * 72];
#pragma unroll
    for (int nt = 0; nt < 4; ++nt)
#pragma unroll
      for (int v = 0; v < 8; ++v)
        pw[(v + 8 * hl) * 72 + nt * 16 + cl] = (__bf16)sc[nt][v];

    // O += P @ V
#pragma unroll
    for (int kk = 0; kk < 2; ++kk) {
      v16bf ap = frag_a16x32(pw, 72, 0, kk * 32, lane);
#pragma unroll
      for (int dt = 0; dt < 4; ++dt) {
        v16bf bv = frag_b32x16(vs, 72, dt * 16, kk * 32, lane);
        accO[dt] = wmma_bf16(ap, bv, accO[dt]);
      }
    }
  }

  // finalize: O / l, store bf16 [B*S, D] with head-major columns
#pragma unroll
  for (int dt = 0; dt < 4; ++dt) {
#pragma unroll
    for (int v = 0; v < 8; ++v) {
      const int qrow = q0 + wave * 16 + v + 8 * hl;
      const int dc = dt * 16 + cl;
      att[(size_t)(b * S_LEN + qrow) * DM + h * DH + dc] =
          (__bf16)(accO[dt][v] / li[v]);
    }
  }
}

// ---------------------------------------------------------------------------
// Launch
// ---------------------------------------------------------------------------
extern "C" void kernel_launch(void* const* d_in, const int* in_sizes, int n_in,
                              void* d_out, int out_size, void* d_ws, size_t ws_size,
                              hipStream_t stream) {
  (void)in_sizes; (void)n_in; (void)out_size; (void)ws_size;
  const float* hidden = (const float*)d_in[0];
  const int* amask    = (const int*)d_in[1];
  const float* wq = (const float*)d_in[2];
  const float* bq = (const float*)d_in[3];
  const float* wk = (const float*)d_in[4];
  const float* bk = (const float*)d_in[5];
  const float* wv = (const float*)d_in[6];
  const float* bv = (const float*)d_in[7];
  const float* wo = (const float*)d_in[8];
  const float* bo = (const float*)d_in[9];
  const float* rel_bias = (const float*)d_in[10];
  const float* ln1g = (const float*)d_in[11];
  const float* ln1b = (const float*)d_in[12];
  const float* ln2g = (const float*)d_in[13];
  const float* ln2b = (const float*)d_in[14];
  const float* w1 = (const float*)d_in[15];
  const float* b1 = (const float*)d_in[16];
  const float* w2 = (const float*)d_in[17];
  const float* b2 = (const float*)d_in[18];
  float* out = (float*)d_out;

  char* ws = (char*)d_ws;
  size_t off = 0;
  auto take = [&](size_t bytes) -> char* {
    char* p = ws + off;
    off += (bytes + 255) & ~(size_t)255;
    return p;
  };
  __bf16* wqT  = (__bf16*)take((size_t)DM * DM * 2);
  __bf16* wkT  = (__bf16*)take((size_t)DM * DM * 2);
  __bf16* wvT  = (__bf16*)take((size_t)DM * DM * 2);
  __bf16* woT  = (__bf16*)take((size_t)DM * DM * 2);
  __bf16* w1T  = (__bf16*)take((size_t)DM * DFF * 2);
  __bf16* w2T  = (__bf16*)take((size_t)DFF * DM * 2);
  __bf16* xln1 = (__bf16*)take((size_t)MROWS * DM * 2);
  __bf16* qbuf = (__bf16*)take((size_t)MROWS * DM * 2);
  __bf16* kbuf = (__bf16*)take((size_t)MROWS * DM * 2);
  __bf16* vbuf = (__bf16*)take((size_t)MROWS * DM * 2);
  __bf16* vtb  = (__bf16*)take((size_t)MROWS * DM * 2);
  __bf16* attb = (__bf16*)take((size_t)MROWS * DM * 2);
  float*  hmid = (float*)take((size_t)MROWS * DM * 4);
  __bf16* xln2 = (__bf16*)take((size_t)MROWS * DM * 2);
  __bf16* f1   = (__bf16*)take((size_t)MROWS * DFF * 2);

  // weight convert+transpose to bf16 [N][K]
  transpose_f32_to_bf16<<<dim3(DM / 16, DM / 16), 256, 0, stream>>>(wq, wqT, DM, DM);
  transpose_f32_to_bf16<<<dim3(DM / 16, DM / 16), 256, 0, stream>>>(wk, wkT, DM, DM);
  transpose_f32_to_bf16<<<dim3(DM / 16, DM / 16), 256, 0, stream>>>(wv, wvT, DM, DM);
  transpose_f32_to_bf16<<<dim3(DM / 16, DM / 16), 256, 0, stream>>>(wo, woT, DM, DM);
  transpose_f32_to_bf16<<<dim3(DM / 16, DFF / 16), 256, 0, stream>>>(w1, w1T, DM, DFF);
  transpose_f32_to_bf16<<<dim3(DFF / 16, DM / 16), 256, 0, stream>>>(w2, w2T, DFF, DM);

  // LN1
  layernorm_to_bf16<<<MROWS, 256, 0, stream>>>(hidden, ln1g, ln1b, xln1);

  // Q,K,V projections
  gemm_bf16_wmma<0><<<dim3(DM / 128, MROWS / 128), 256, 0, stream>>>(
      xln1, wqT, bq, nullptr, nullptr, qbuf, nullptr, MROWS, DM, DM);
  gemm_bf16_wmma<0><<<dim3(DM / 128, MROWS / 128), 256, 0, stream>>>(
      xln1, wkT, bk, nullptr, nullptr, kbuf, nullptr, MROWS, DM, DM);
  gemm_bf16_wmma<0><<<dim3(DM / 128, MROWS / 128), 256, 0, stream>>>(
      xln1, wvT, bv, nullptr, nullptr, vbuf, nullptr, MROWS, DM, DM);

  // V -> [B,H,Dh,S] for direct WMMA-B staging in attention
  transpose_v_bf16<<<dim3(S_LEN / 16, DH / 16, NB * NH), 256, 0, stream>>>(vbuf, vtb);

  // attention
  attn_flash_wmma<<<dim3(S_LEN / 64, NB * NH), 128, 0, stream>>>(
      qbuf, kbuf, vtb, rel_bias, amask, attb);

  // output projection + mask + residual -> h (f32)
  gemm_bf16_wmma<2><<<dim3(DM / 128, MROWS / 128), 256, 0, stream>>>(
      attb, woT, bo, hidden, amask, nullptr, hmid, MROWS, DM, DM);

  // LN2
  layernorm_to_bf16<<<MROWS, 256, 0, stream>>>(hmid, ln2g, ln2b, xln2);

  // FFN1 + GELU
  gemm_bf16_wmma<1><<<dim3(DFF / 128, MROWS / 128), 256, 0, stream>>>(
      xln2, w1T, b1, nullptr, nullptr, f1, nullptr, MROWS, DFF, DM);

  // FFN2 + residual + mask -> out (f32)
  gemm_bf16_wmma<3><<<dim3(DM / 128, MROWS / 128), 256, 0, stream>>>(
      f1, w2T, b2, hmid, amask, nullptr, out, MROWS, DM, DFF);
}
